// Transformer_54004918780551
// MI455X (gfx1250) — compile-verified, matrix operand order checked by reading, not run on hardware
//
#include <hip/hip_runtime.h>
#include <math.h>

typedef _Float16 f16;
typedef __attribute__((ext_vector_type(16))) _Float16 v16h;
typedef __attribute__((ext_vector_type(8)))  float    v8f;

#define B_   2
#define C_   256
#define T_   16
#define H_   32
#define W_   32
#define L_   4
#define NO_  4
#define N_   512
#define THW_ 16384
#define MROWS_ (B_*L_*N_)            // 4096
#define OUT_ELEMS (B_*C_*THW_*L_)    // 33554432
#define NTOT_ 256                    // output columns of both GEMMs

__constant__ int c_tg[8] = {1,3,5,7,8,10,12,14};      // round(linspace(1,14,8))
__constant__ int c_hw[8] = {2,6,10,14,17,21,25,29};   // round(linspace(2,29,8))

__device__ __forceinline__ int flat_idx(int n) {
    return c_tg[n >> 6] * (H_*W_) + c_hw[(n >> 3) & 7] * W_ + c_hw[n & 7];
}
__device__ __forceinline__ float sigm(float x) { return 1.f / (1.f + expf(-x)); }
__device__ __forceinline__ float invsig(float x) {
    x = fminf(fmaxf(x, 0.f), 1.f);
    return logf(fmaxf(x, 1e-5f) / fmaxf(1.f - x, 1e-5f));
}
// fragment element i (0..15) -> k offset within a 32-wide k-tile, given lane half
__device__ __forceinline__ int frag_k(int q, int hid) {
    return ((q >> 3) << 4) + hid * 8 + (q & 7);
}

// ---------------- global mean over THW per (b,c): float4 over L ----------------
__global__ __launch_bounds__(256) void k_glob(const float* __restrict__ feat,
                                              float* __restrict__ glob) {
    __shared__ float4 red[256];
    int bc = blockIdx.x;                  // b*C + c
    int t = threadIdx.x;
    const float4* p = (const float4*)(feat + (size_t)bc * THW_ * L_);
    float4 s = make_float4(0.f, 0.f, 0.f, 0.f);
    for (int thw = t; thw < THW_; thw += 256) {
        float4 v = p[thw];
        s.x += v.x; s.y += v.y; s.z += v.z; s.w += v.w;
    }
    red[t] = s; __syncthreads();
    for (int st = 128; st > 0; st >>= 1) {
        if (t < st) {
            red[t].x += red[t+st].x; red[t].y += red[t+st].y;
            red[t].z += red[t+st].z; red[t].w += red[t+st].w;
        }
        __syncthreads();
    }
    if (t < 4) glob[bc * L_ + t] = (&red[0].x)[t] * (1.f / THW_);
}

// ---------------- fill output with ln_b[c] ----------------
__global__ __launch_bounds__(256) void k_fill(float* __restrict__ out,
                                              const float* __restrict__ ln_b) {
    int gid = blockIdx.x * 256 + threadIdx.x;     // one float4 per thread
    int c = (gid >> 14) & 255;                    // (4*gid)/(THW*L) % C
    float v = ln_b[c];
    reinterpret_cast<float4*>(out)[gid] = make_float4(v, v, v, v);
}

// ---------------- build src directly in WMMA-A fragment-packed layout ----------------
// Apack[((kt*M + m)*2 + hid)*16 + q],  k = kt*32 + frag_k(q,hid)
__global__ __launch_bounds__(256) void k_src(const float* __restrict__ feat,
                                             const float* __restrict__ pos,
                                             const float* __restrict__ glob,
                                             f16* __restrict__ Apack) {
    int dst = blockIdx.x * 256 + threadIdx.x;
    int q   = dst & 15;
    int hid = (dst >> 4) & 1;
    int m   = (dst >> 5) & (MROWS_ - 1);
    int kt  = dst >> 17;                           // 5 + log2(4096)
    int kdim = kt * 32 + frag_k(q, hid);
    int b = m / (L_ * N_);
    int rem = m % (L_ * N_);
    int l = rem / N_, n = rem % N_;
    float v;
    if (kdim < 512) {
        int c = kdim & 255;
        const float* p = (kdim < 256) ? feat : pos;
        v = p[((size_t)(b * C_ + c) * THW_ + flat_idx(n)) * L_ + l];
    } else {
        v = glob[(b * C_ + (kdim - 512)) * L_ + l];
    }
    Apack[dst] = (f16)v;
}

// ---------------- pack weight matrix (K x 256 f32) into WMMA-B fragment layout ----------------
// Bpack[((kt*256 + n)*2 + hid)*16 + q]
__global__ __launch_bounds__(256) void k_packB(const float* __restrict__ Bw,
                                               f16* __restrict__ Bpack) {
    int dst = blockIdx.x * 256 + threadIdx.x;
    int q   = dst & 15;
    int hid = (dst >> 4) & 1;
    int n   = (dst >> 5) & (NTOT_ - 1);
    int kt  = dst >> 13;                           // 5 + log2(256)
    int k   = kt * 32 + frag_k(q, hid);
    Bpack[dst] = (f16)Bw[(size_t)k * NTOT_ + n];
}

// ---------------- WMMA GEMM, fully unrolled, no LDS ----------------
// out = relu(A @ B + bias); A/B consumed via packed fragment loads (2x b128 each).
template <int KT, bool OUTH>
__global__ __launch_bounds__(256) void k_gemm(const f16* __restrict__ Apack,
                                              const f16* __restrict__ Bpack,
                                              const float* __restrict__ bias,
                                              f16* __restrict__ outh,
                                              float* __restrict__ outf) {
    int tid  = threadIdx.x;
    int m0   = blockIdx.y * 16;
    int wave = tid >> 5, lane = tid & 31;
    int hid  = lane >> 4, r = lane & 15;
    int nglob = blockIdx.x * 128 + wave * 16 + r;
    const f16* ap = Apack + ((size_t)(m0 + r) * 2 + hid) * 16;
    const f16* bp = Bpack + ((size_t)nglob * 2 + hid) * 16;
    const size_t astride = (size_t)MROWS_ * 32;   // halfs per k-tile
    const size_t bstride = (size_t)NTOT_  * 32;
    v8f acc = {};
#pragma unroll
    for (int kt = 0; kt < KT; ++kt) {
        v16h a = *(const v16h*)(ap + kt * astride);
        v16h b = *(const v16h*)(bp + kt * bstride);
        acc = __builtin_amdgcn_wmma_f32_16x16x32_f16(false, a, false, b,
                                                     (short)0, acc, false, false);
    }
    float bs = bias[nglob];
#pragma unroll
    for (int j = 0; j < 8; ++j) {
        int mm = m0 + j + 8 * hid;                // ISA C/D layout: m = vgpr + 8*(lane/16)
        float v = acc[j] + bs;
        v = v > 0.f ? v : 0.f;
        if (OUTH) {                               // store in A-pack layout for next GEMM
            int kt2 = nglob >> 5, kl = nglob & 31;
            int hid2 = (kl >> 3) & 1;
            int q2 = ((kl >> 4) << 3) | (kl & 7);
            outh[((size_t)(kt2 * MROWS_ + mm) * 2 + hid2) * 16 + q2] = (f16)v;
        } else {
            outf[(size_t)mm * NTOT_ + nglob] = v;
        }
    }
}

// ---------------- per-sample-point scalars: smp, weight_o, wl_sum, next_ind ----------------
// scal[m*17]: [0..11] smp(g,xyz), [12..15] w_o(g), [16] wl_val
__global__ __launch_bounds__(256) void k_scal(const float* __restrict__ offsrc,
                                              const float* __restrict__ off_w, const float* __restrict__ off_b,
                                              const float* __restrict__ wo_w,  const float* __restrict__ wo_b,
                                              const float* __restrict__ wl_w,  const float* __restrict__ wl_b,
                                              const float* __restrict__ next_w,const float* __restrict__ next_b,
                                              float* __restrict__ scal, int* __restrict__ next_out) {
    __shared__ float red[256];
    int m = blockIdx.x, t = threadIdx.x;
    int n = m % N_;
    int d = t & 63, g = t >> 6;
    float v = offsrc[(size_t)m * C_ + t];

    float nxt[3], wl[4];
    for (int x = 0; x < 3; ++x) {
        red[t] = v * next_w[t * 3 + x]; __syncthreads();
        for (int s = 128; s > 0; s >>= 1) { if (t < s) red[t] += red[t + s]; __syncthreads(); }
        nxt[x] = red[0] + next_b[x]; __syncthreads();
    }
    for (int j = 0; j < 4; ++j) {
        red[t] = v * wl_w[t * 4 + j]; __syncthreads();
        for (int s = 128; s > 0; s >>= 1) { if (t < s) red[t] += red[t + s]; __syncthreads(); }
        wl[j] = red[0] + wl_b[j]; __syncthreads();
    }
    float offv[3], wo0 = 0.f, wo1 = 0.f;
    for (int x = 0; x < 3; ++x) {                 // 64-wide segment reductions, 4 groups in parallel
        red[t] = v * off_w[d * 3 + x]; __syncthreads();
        for (int s = 32; s > 0; s >>= 1) { if (d < s) red[t] += red[t + s]; __syncthreads(); }
        offv[x] = red[g * 64] + off_b[x]; __syncthreads();
    }
    for (int j = 0; j < 2; ++j) {
        red[t] = v * wo_w[d * 2 + j]; __syncthreads();
        for (int s = 32; s > 0; s >>= 1) { if (d < s) red[t] += red[t + s]; __syncthreads(); }
        float rr = red[g * 64] + wo_b[j];
        if (j == 0) wo0 = rr; else wo1 = rr;
        __syncthreads();
    }

    int fi = flat_idx(n);
    float tn = (float)(fi / (H_ * W_)) / (float)(T_ - 1);
    float hn = (float)((fi / W_) % H_) / (float)(H_ - 1);
    float wn = (float)(fi % W_) / (float)(W_ - 1);

    if (d == 0) {
        scal[m * 17 + g * 3 + 0] = sigm(invsig(tn) + offv[0]) * 2.f - 1.f;
        scal[m * 17 + g * 3 + 1] = sigm(invsig(hn) + offv[1]) * 2.f - 1.f;
        scal[m * 17 + g * 3 + 2] = sigm(invsig(wn) + offv[2]) * 2.f - 1.f;
        scal[m * 17 + 12 + g]    = sigm(wo1 - wo0);           // softmax(...)[1]
    }
    if (t == 0) {
        float mx = fmaxf(fmaxf(wl[0], wl[1]), fmaxf(wl[2], wl[3]));
        float e0 = expf(wl[0]-mx), e1 = expf(wl[1]-mx), e2 = expf(wl[2]-mx), e3 = expf(wl[3]-mx);
        float S = e0 + e1 + e2 + e3;
        scal[m * 17 + 16] = e0/S + e1/S + e2/S + e3/S;        // wl row-sum, computed faithfully
        int di0 = (int)rintf(sigm(invsig(tn) + nxt[0] * 10.f) * (float)(T_ - 1));
        int di1 = (int)rintf(sigm(invsig(hn) + nxt[1] * 10.f) * (float)(H_ - 1));
        int di2 = (int)rintf(sigm(invsig(wn) + nxt[2] * 10.f) * (float)(W_ - 1));
        next_out[m] = (H_ * W_) * di0 + W_ * di1 + di2;
    }
}

// ---------------- final: trilinear gather + residual + wl + LayerNorm at sampled voxels ----------------
__global__ __launch_bounds__(256) void k_final(const float* __restrict__ feat,
                                               const float* __restrict__ scal,
                                               const float* __restrict__ ln_g,
                                               const float* __restrict__ ln_b,
                                               float* __restrict__ out) {
    __shared__ float red[256];
    int m = blockIdx.x, c = threadIdx.x;
    int n = m % N_;
    int l = (m / N_) % L_;
    int b = m / (N_ * L_);
    int fi = flat_idx(n);
    size_t base = (size_t)(b * C_ + c) * THW_ * L_ + l;
    float acc = feat[base + (size_t)fi * L_];
    for (int g = 0; g < NO_; ++g) {
        float s0 = scal[m*17 + g*3 + 0];  // t-derived coord -> W axis (reference wiring)
        float s1 = scal[m*17 + g*3 + 1];  // h-derived -> H axis
        float s2 = scal[m*17 + g*3 + 2];  // w-derived -> T axis
        float wo = scal[m*17 + 12 + g];
        float ix = ((s0 + 1.f) * W_ - 1.f) * 0.5f;
        float iy = ((s1 + 1.f) * H_ - 1.f) * 0.5f;
        float iz = ((s2 + 1.f) * T_ - 1.f) * 0.5f;
        float x0 = floorf(ix), y0 = floorf(iy), z0 = floorf(iz);
        float sv = 0.f;
        for (int dz = 0; dz < 2; ++dz)
        for (int dy = 0; dy < 2; ++dy)
        for (int dx = 0; dx < 2; ++dx) {
            float xc = x0 + dx, yc = y0 + dy, zc = z0 + dz;
            if (xc < 0.f || xc > (float)(W_-1) || yc < 0.f || yc > (float)(H_-1) ||
                zc < 0.f || zc > (float)(T_-1)) continue;
            float wgt = (1.f - fabsf(ix - xc)) * (1.f - fabsf(iy - yc)) * (1.f - fabsf(iz - zc));
            int idx = (((int)zc) * H_ + (int)yc) * W_ + (int)xc;
            sv += wgt * feat[base + (size_t)idx * L_];
        }
        acc += wo * sv;
    }
    acc *= scal[m*17 + 16];
    red[c] = acc; __syncthreads();
    for (int s = 128; s > 0; s >>= 1) { if (c < s) red[c] += red[c + s]; __syncthreads(); }
    float mu = red[0] * (1.f / C_); __syncthreads();
    float df = acc - mu;
    red[c] = df * df; __syncthreads();
    for (int s = 128; s > 0; s >>= 1) { if (c < s) red[c] += red[c + s]; __syncthreads(); }
    float var = red[0] * (1.f / C_);
    out[base + (size_t)fi * L_] = df / sqrtf(var + 1e-5f) * ln_g[c] + ln_b[c];
}

extern "C" void kernel_launch(void* const* d_in, const int* in_sizes, int n_in,
                              void* d_out, int out_size, void* d_ws, size_t ws_size,
                              hipStream_t stream) {
    const float* features = (const float*)d_in[0];
    const float* pos      = (const float*)d_in[1];
    const float* proj_w1  = (const float*)d_in[2];
    const float* proj_b1  = (const float*)d_in[3];
    const float* proj_w2  = (const float*)d_in[4];
    const float* proj_b2  = (const float*)d_in[5];
    const float* off_w    = (const float*)d_in[6];
    const float* off_b    = (const float*)d_in[7];
    const float* wo_w     = (const float*)d_in[8];
    const float* wo_b     = (const float*)d_in[9];
    const float* wl_w     = (const float*)d_in[10];
    const float* wl_b     = (const float*)d_in[11];
    const float* next_w   = (const float*)d_in[12];
    const float* next_b   = (const float*)d_in[13];
    const float* ln_g     = (const float*)d_in[14];
    const float* ln_b     = (const float*)d_in[15];

    char* ws = (char*)d_ws;
    f16*   Apack  = (f16*)(ws);                      // 4096*768*2        = 6291456
    f16*   Bpack1 = (f16*)(ws + 6291456);            // 768*256*2         = 393216
    f16*   Bpack2 = (f16*)(ws + 6684672);            // 256*256*2         = 131072
    f16*   h1pack = (f16*)(ws + 6815744);            // 4096*256*2        = 2097152
    float* offsrc = (float*)(ws + 8912896);          // 4096*256*4        = 4194304
    float* glob   = (float*)(ws + 13107200);         // 2048*4            = 8192
    float* scal   = (float*)(ws + 13115392);         // 4096*17*4         = 278528

    float* out_f = (float*)d_out;
    int*   next_out = (int*)(out_f + OUT_ELEMS);

    k_glob<<<B_ * C_, 256, 0, stream>>>(features, glob);
    k_fill<<<OUT_ELEMS / 4 / 256, 256, 0, stream>>>(out_f, ln_b);
    k_src<<<(MROWS_ * 768) / 256, 256, 0, stream>>>(features, pos, glob, Apack);
    k_packB<<<(768 * NTOT_) / 256, 256, 0, stream>>>(proj_w1, Bpack1);
    k_packB<<<(256 * NTOT_) / 256, 256, 0, stream>>>(proj_w2, Bpack2);
    k_gemm<24, true ><<<dim3(NTOT_ / 128, MROWS_ / 16), 256, 0, stream>>>(Apack, Bpack1, proj_b1,
                                                                          h1pack, nullptr);
    k_gemm< 8, false><<<dim3(NTOT_ / 128, MROWS_ / 16), 256, 0, stream>>>(h1pack, Bpack2, proj_b2,
                                                                          nullptr, offsrc);
    k_scal<<<MROWS_, 256, 0, stream>>>(offsrc, off_w, off_b, wo_w, wo_b,
                                       wl_w, wl_b, next_w, next_b, scal, next_out);
    k_final<<<MROWS_, 256, 0, stream>>>(features, scal, ln_g, ln_b, out_f);
}
